// DiagonalVariance_60309930770997
// MI455X (gfx1250) — matrix operand order repaired and checked by reading, not compiled, not run on hardware
//
#include <hip/hip_runtime.h>
#include <hip/hip_bf16.h>

// ---------------------------------------------------------------------------
// Fused per-dim MLP: out[b,d] = softplus( L3( softplus(L2( softplus(L1(x)) ))))
//   x = [y[b,d], t[b,0..2]]  (4 wide),  L1: 4->64, L2: 64->64, L3: 64->1
//
// Strategy (MI455X / gfx1250, wave32):
//  * Compute transposed: Cᵀ = Wᵀ · Xᵀ so weights sit in the WMMA A operand
//    (M = hidden-out across lanes) and activations in the B operand
//    (N = batch row across lanes, K packed per lane).
//  * Layer 1: V_WMMA_F32_16X16X4_F32  (K=4 == MLP input width, full f32).
//  * Layer 2: V_WMMA_F32_16X16X32_F16 (f16 inputs, f32 accumulate) — ~93% of
//    the FLOPs ride the fast f16 matrix pipe.
//  * Layer-1 C-layout -> layer-2 B-layout is a lane-local f16 repack:
//    no shuffle, no LDS transpose at all.
//  * W1ᵀ/W2ᵀ tiles loaded into registers ONCE per block (L2-resident).
//  * Softplus in native base-2 (v_exp_f32/v_log_f32 are exp2/log2): 2 TRANS
//    + ~4 VALU per value; relu term via single v_med3_f32.
// ---------------------------------------------------------------------------

typedef __attribute__((ext_vector_type(2)))  float    v2f;
typedef __attribute__((ext_vector_type(8)))  float    v8f;
typedef __attribute__((ext_vector_type(16))) _Float16 v16h;

#define DVN_D   16
#define DVN_TE  3
#define DVN_H   64
#define DVN_ITER 16                      // row-tiles per wave per block
#define DVN_ROWS_PER_BLOCK (16 * 8 * DVN_ITER)   // 2048

__device__ __forceinline__ float dvn_softplus(float x) {
    // softplus(x) = max(x,0) + ln2 * log2(1 + exp2(-|x| * log2e))
    // v_exp_f32 / v_log_f32 are natively base-2 on gfx1250.
    // max(x,0) as med3(x, 0, +inf): one v_med3_f32, no canonicalize pre-op.
    const float LOG2E = 1.44269504088896340736f;
    const float LN2   = 0.69314718055994530942f;
    float e    = __builtin_amdgcn_exp2f(-fabsf(x) * LOG2E);
    float l    = __builtin_amdgcn_logf(1.0f + e);   // log2(1+e); arg in [1,2]
    float relu = __builtin_amdgcn_fmed3f(x, 0.0f, __builtin_inff());
    return fmaf(LN2, l, relu);
}

__global__ __launch_bounds__(256)
void DiagonalVariance_60309930770997_kernel(
        const float* __restrict__ t,   // [B, 3]
        const float* __restrict__ y,   // [B, 16]
        const float* __restrict__ W1,  // [D, 4, 64]
        const float* __restrict__ b1,  // [D, 64]
        const float* __restrict__ W2,  // [D, 64, 64]  (hin, hout)
        const float* __restrict__ b2,  // [D, 64]
        const float* __restrict__ W3,  // [D, 64, 1]
        const float* __restrict__ b3,  // [D, 1]
        float* __restrict__ out)       // [B, 16]
{
    const int d    = blockIdx.y;
    const int lane = threadIdx.x & 31;
    const int wave = threadIdx.x >> 5;
    const int half = lane >> 4;        // 0: lanes 0-15, 1: lanes 16-31
    const int lid  = lane & 15;
    const bool h0  = (half == 0);

    // --- stage small per-dim vectors in LDS (broadcast reads later) --------
    __shared__ float s_b1[DVN_H], s_b2[DVN_H], s_w3[DVN_H];
    __shared__ float s_b3;
    for (int i = threadIdx.x; i < DVN_H; i += 256) {
        s_b1[i] = b1[d * DVN_H + i];
        s_b2[i] = b2[d * DVN_H + i];
        s_w3[i] = W3[d * DVN_H + i];
    }
    if (threadIdx.x == 0) s_b3 = b3[d];
    __syncthreads();

    // --- A operand = W1ᵀ tiles, f32 16x16x4 layout -------------------------
    // lane<16:  M = mo*16+lane,    v0=K0, v1=K1
    // lane>=16: M = mo*16+lane-16, v0=K2, v1=K3
    v2f a1[4];
    {
        const float* w1d = W1 + (size_t)d * 4 * DVN_H;
        #pragma unroll
        for (int mo = 0; mo < 4; ++mo) {
            const int hout = mo * 16 + lid;
            a1[mo].x = w1d[(half * 2 + 0) * DVN_H + hout];
            a1[mo].y = w1d[(half * 2 + 1) * DVN_H + hout];
        }
    }

    // --- A operand = W2ᵀ tiles, f16 16x16x32 layout ------------------------
    // lane<16:  M = mo*16+lane,    elems 0..7 -> K=kb+0..7,  8..15 -> K=kb+16..23 (kb=kt*32)
    // lane>=16: M = mo*16+lane-16, elems 0..7 -> K=kb+8..15, 8..15 -> K=kb+24..31
    v16h a2[4][2];
    {
        const float* w2d = W2 + (size_t)d * DVN_H * DVN_H;   // [hin][hout]
        #pragma unroll
        for (int mo = 0; mo < 4; ++mo) {
            const int hout = mo * 16 + lid;
            #pragma unroll
            for (int kt = 0; kt < 2; ++kt) {
                const int kb = kt * 32 + half * 8;
                #pragma unroll
                for (int i = 0; i < 8; ++i) {
                    a2[mo][kt][i]     = (_Float16)w2d[(kb + i)      * DVN_H + hout];
                    a2[mo][kt][8 + i] = (_Float16)w2d[(kb + 16 + i) * DVN_H + hout];
                }
            }
        }
    }

    const size_t rowbase = (size_t)blockIdx.x * DVN_ROWS_PER_BLOCK;

    for (int it = 0; it < DVN_ITER; ++it) {
        const size_t row0 = rowbase + (size_t)(it * 8 + wave) * 16;
        const size_t row  = row0 + lid;

        // prefetch next tile of streaming inputs (global_prefetch_b8)
        __builtin_prefetch(&y[(row + 128) * DVN_D + d], 0, 1);
        __builtin_prefetch(&t[(row + 128) * DVN_TE],     0, 1);

        // --- B operand = Xᵀ, f32 4x16 layout (branchless address select) ---
        // lane<16:  N=row, v0=K0=y, v1=K1=t0 ; lane>=16: N=row, v0=K2=t1, v1=K3=t2
        const float* py = y + row * DVN_D + d;
        const float* pt = t + row * DVN_TE;
        const float* p0 = h0 ? py : (pt + 1);
        const float* p1 = h0 ? pt : (pt + 2);
        v2f bx;
        bx.x = *p0;
        bx.y = *p1;

        // --- layer 1: h1ᵀ = W1ᵀ·xᵀ + b1, 4 M-tiles of 16, K=4 --------------
        // C layout: VGPR r -> M = mo*16 + r + half*8 ; lane -> N = row
        v8f c1[4];
        #pragma unroll
        for (int mo = 0; mo < 4; ++mo) {
            #pragma unroll
            for (int r = 0; r < 8; ++r)
                c1[mo][r] = s_b1[mo * 16 + r + half * 8];   // bias preloaded into C
            c1[mo] = __builtin_amdgcn_wmma_f32_16x16x4_f32(
                         false, a1[mo], false, bx, (short)0, c1[mo], false, false);
            #pragma unroll
            for (int r = 0; r < 8; ++r)
                c1[mo][r] = dvn_softplus(c1[mo][r]);
        }

        // --- layer 2: h2ᵀ = W2ᵀ·h1ᵀ + b2 -----------------------------------
        v8f c2[4];
        #pragma unroll
        for (int mo = 0; mo < 4; ++mo)
            #pragma unroll
            for (int r = 0; r < 8; ++r)
                c2[mo][r] = s_b2[mo * 16 + r + half * 8];

        #pragma unroll
        for (int kt = 0; kt < 2; ++kt) {
            // Repack c1 (C layout) into B f16 32x16 layout for this K-tile:
            // lane needs K = kt*32 + half*8 + e (elems 0..7) and
            // kt*32 + 16 + half*8 + e (elems 8..15): all local, no shuffle.
            v16h bh;
            #pragma unroll
            for (int e = 0; e < 8; ++e) {
                bh[e]     = (_Float16)c1[2 * kt][e];
                bh[e + 8] = (_Float16)c1[2 * kt + 1][e];
            }
            #pragma unroll
            for (int mo2 = 0; mo2 < 4; ++mo2)
                c2[mo2] = __builtin_amdgcn_wmma_f32_16x16x32_f16(
                              false, a2[mo2][kt], false, bh, (short)0, c2[mo2],
                              false, false);
        }

        // --- layer 3: out = softplus( h2 · W3 + b3 ) -----------------------
        // Each lane covers h = mo*16 + r + half*8; partner lane covers the
        // complementary 32 of the 64 hidden units.
        float partial = 0.0f;
        #pragma unroll
        for (int mo = 0; mo < 4; ++mo)
            #pragma unroll
            for (int r = 0; r < 8; ++r) {
                const float h2 = dvn_softplus(c2[mo][r]);
                partial = fmaf(h2, s_w3[mo * 16 + r + half * 8], partial);
            }
        const float total = partial + __shfl_xor(partial, 16, 32);
        if (h0) {
            out[row * DVN_D + d] = dvn_softplus(total + s_b3);
        }
    }
}

extern "C" void kernel_launch(void* const* d_in, const int* in_sizes, int n_in,
                              void* d_out, int out_size, void* d_ws, size_t ws_size,
                              hipStream_t stream) {
    const float* t  = (const float*)d_in[0];
    const float* y  = (const float*)d_in[1];
    const float* W1 = (const float*)d_in[2];
    const float* b1 = (const float*)d_in[3];
    const float* W2 = (const float*)d_in[4];
    const float* b2 = (const float*)d_in[5];
    const float* W3 = (const float*)d_in[6];
    const float* b3 = (const float*)d_in[7];
    float* out = (float*)d_out;

    const int B = in_sizes[1] / DVN_D;               // y is [B, 16]
    dim3 grid(B / DVN_ROWS_PER_BLOCK, DVN_D);        // (128, 16) for B=262144
    DiagonalVariance_60309930770997_kernel<<<grid, 256, 0, stream>>>(
        t, y, W1, b1, W2, b2, W3, b3, out);
}
